// VIT_76089640616631
// MI455X (gfx1250) — matrix operand order repaired and checked by reading, not compiled, not run on hardware
//
#include <hip/hip_runtime.h>
#include <hip/hip_bf16.h>

#define B_ 32
#define C_ 3
#define IMG_ 224
#define NP_ 14
#define PS_ 16
#define IN_DIM_ 588
#define INPAD_ 608            // IN_DIM_ padded to multiple of 32
#define D_ 768
#define NH_ 12
#define DH_ 64
#define L_ 12
#define MLP_ 3072
#define OUT_ 1000
#define OUTPAD_ 1024          // OUT_ padded to multiple of 64
#define S_ 197
#define TOK_ 6304             // B_*S_
#define TOKPAD_ 6400          // padded to multiple of 128
#define EPS_ 1e-5f

typedef __attribute__((ext_vector_type(16))) __bf16 v16bf;
typedef __attribute__((ext_vector_type(8)))  float  v8f;
typedef __attribute__((ext_vector_type(8)))  unsigned int v8u;

union FragBF { v8u u; v16bf bf; };

__device__ __forceinline__ unsigned short f2bf(float f) {
  unsigned int u = __float_as_uint(f);
  unsigned int r = u + 0x7FFFu + ((u >> 16) & 1u);  // round-to-nearest-even
  return (unsigned short)(r >> 16);
}
__device__ __forceinline__ float gelu_exact(float v) {
  return 0.5f * v * (1.0f + erff(v * 0.70710678118654752f));
}
__device__ __forceinline__ int afrag_kk(int p, int hf) {
  return (p < 4) ? (hf * 8 + 2 * p) : (16 + hf * 8 + 2 * (p - 4));
}

// gfx1250 async global->LDS copy (16B per lane), tracked on ASYNCcnt.
__device__ __forceinline__ void async_ld_b128(const unsigned short* g, unsigned short* l) {
  unsigned lds = (unsigned)(unsigned long long)(uintptr_t)l;   // as(3) offset = low 32 bits
  unsigned long long ga = (unsigned long long)(uintptr_t)g;
  asm volatile("global_load_async_to_lds_b128 %0, %1, off" :: "v"(lds), "v"(ga) : "memory");
}
__device__ __forceinline__ void wait_async0() {
  asm volatile("s_wait_asynccnt 0x0" ::: "memory");
}

// ---------------------------------------------------------------------------
// bf16-native WMMA GEMM: C = act(A @ B^T(stored N-major) + bias) [+ residual]
// A: bf16 [Mpad][lda] (K contiguous), B: bf16 [Npad][ldb] (K contiguous).
// Tiles staged with GLOBAL_LOAD_ASYNC_TO_LDS_B128 (no predication: buffers padded,
// K multiple of 32, Npad multiple of 64, A rows padded to m-block multiple).
// Block 256 threads / 8 waves; tile 128(M) x 64(N); K-step 32; wave = 32x32.
// ---------------------------------------------------------------------------
__global__ __launch_bounds__(256) void vit_wmma_gemm_bf(
    const unsigned short* __restrict__ A, long long lda, long long bsA,
    const unsigned short* __restrict__ Bm, long long ldb, long long bsB,
    const float* __restrict__ bias, long long bsBias,
    const float* __restrict__ residual,
    void* __restrict__ Cm, long long ldc, long long bsC, int cIsBf16,
    int M, int N, int K, int act)
{
  __shared__ __align__(16) unsigned short As[128 * 32];  // [m][k]
  __shared__ __align__(16) unsigned short Bs[64 * 32];   // [n][k]

  const int tid  = threadIdx.x;
  const int lane = tid & 31;
  const int warp = tid >> 5;
  const int wm   = warp >> 1;
  const int wn   = warp & 1;
  const int hf   = lane >> 4;
  const int m16  = lane & 15;

  const int z = blockIdx.z;
  A  += (size_t)z * bsA;
  Bm += (size_t)z * bsB;
  if (bias)     bias     += (size_t)z * bsBias;
  const size_t coff = (size_t)z * bsC;

  const int m0 = blockIdx.y * 128;
  const int n0 = blockIdx.x * 64;

  v8f acc[2][2] = {};

  for (int k0 = 0; k0 < K; k0 += 32) {
    // B tile: 64 rows x 64B = 256 chunks (1/thread)
    {
      int brow = tid >> 2, bch = tid & 3;
      async_ld_b128(&Bm[(size_t)(n0 + brow) * ldb + k0 + bch * 8], &Bs[tid * 8]);
    }
    // A tile: 128 rows x 64B = 512 chunks (2/thread)
    #pragma unroll
    for (int i = 0; i < 2; ++i) {
      int cid = tid + i * 256;
      int arow = cid >> 2, ach = cid & 3;
      async_ld_b128(&A[(size_t)(m0 + arow) * lda + k0 + ach * 8], &As[cid * 8]);
    }
    wait_async0();
    __syncthreads();

    FragBF fa[2], fb[2];
    #pragma unroll
    for (int r = 0; r < 2; ++r) {
      int row = wm * 32 + r * 16 + m16;
      #pragma unroll
      for (int p = 0; p < 8; ++p)
        fa[r].u[p] = *(const unsigned int*)&As[row * 32 + afrag_kk(p, hf)];
    }
    #pragma unroll
    for (int c = 0; c < 2; ++c) {
      int col = wn * 32 + c * 16 + m16;
      #pragma unroll
      for (int p = 0; p < 8; ++p)
        fb[c].u[p] = *(const unsigned int*)&Bs[col * 32 + afrag_kk(p, hf)];
    }
    #pragma unroll
    for (int r = 0; r < 2; ++r)
      #pragma unroll
      for (int c = 0; c < 2; ++c)
        acc[r][c] = __builtin_amdgcn_wmma_f32_16x16x32_bf16(
            false, fa[r].bf, false, fb[c].bf, (short)0, acc[r][c], false, false);
    __syncthreads();
  }

  #pragma unroll
  for (int r = 0; r < 2; ++r) {
    #pragma unroll
    for (int c = 0; c < 2; ++c) {
      int n = n0 + wn * 32 + c * 16 + m16;
      #pragma unroll
      for (int i = 0; i < 8; ++i) {
        int m = m0 + wm * 32 + r * 16 + hf * 8 + i;
        if (m < M && n < N) {
          float v = acc[r][c][i];
          if (bias) v += bias[n];
          if (act == 1) v = gelu_exact(v);
          if (residual) v += residual[coff + (size_t)m * ldc + n];
          if (cIsBf16)
            ((unsigned short*)Cm)[coff + (size_t)m * ldc + n] = f2bf(v);
          else
            ((float*)Cm)[coff + (size_t)m * ldc + n] = v;
        }
      }
    }
  }
}

// ---------------------------------------------------------------------------
// WMMA flash-attention on bf16 q/k/v; fp32 residual accumulate into out.
// Block = (128-query tile, h, b); 8 waves x 16 q-rows. Q/K staged async.
// ---------------------------------------------------------------------------
__global__ __launch_bounds__(256) void vit_attention_wmma(
    const unsigned short* __restrict__ q, const unsigned short* __restrict__ k,
    const unsigned short* __restrict__ v, float* __restrict__ out)
{
  __shared__ __align__(16) unsigned short qs[128 * 64];
  __shared__ __align__(16) unsigned short ks[32 * 64];
  __shared__ __align__(16) unsigned short vsT[64 * 32];   // [e][key]
  __shared__ __align__(16) unsigned short plds[8 * 16 * 32];

  const int tid  = threadIdx.x;
  const int lane = tid & 31;
  const int wid  = tid >> 5;
  const int hf   = lane >> 4;
  const int m16  = lane & 15;

  const int q0 = blockIdx.x * 128;
  const int h  = blockIdx.y;
  const int b  = blockIdx.z;
  const size_t base = ((size_t)b * S_) * D_ + (size_t)h * DH_;
  const float scale = 0.125f;

  // stage Q block async: 128 rows x 128B = 1024 chunks (4/thread); rows clamped
  #pragma unroll
  for (int i = 0; i < 4; ++i) {
    int cid = tid + i * 256;
    int r = cid >> 3, ch = cid & 7;
    int gq = q0 + r; if (gq >= S_) gq = S_ - 1;
    async_ld_b128(&q[base + (size_t)gq * D_ + ch * 8], &qs[cid * 8]);
  }
  wait_async0();
  __syncthreads();

  FragBF qa[2];
  #pragma unroll
  for (int c0 = 0; c0 < 2; ++c0)
    #pragma unroll
    for (int p = 0; p < 8; ++p)
      qa[c0].u[p] = *(const unsigned int*)
          &qs[(wid * 16 + m16) * 64 + c0 * 32 + afrag_kk(p, hf)];

  float m_run[8], l_run[8];
  v8f oacc[4] = {};
  #pragma unroll
  for (int i = 0; i < 8; ++i) { m_run[i] = -3.4e38f; l_run[i] = 0.f; }

  const int pbase = wid * 16 * 32;

  for (int kt = 0; kt < S_; kt += 32) {
    // K tile async: 32 rows x 128B = 256 chunks (1/thread)
    {
      int r = tid >> 3, ch = tid & 7;
      int gk = kt + r; if (gk >= S_) gk = S_ - 1;
      async_ld_b128(&k[base + (size_t)gk * D_ + ch * 8], &ks[tid * 8]);
    }
    // V^T tile: transpose scatter (bf16 pair loads, u16 LDS stores)
    #pragma unroll
    for (int i = 0; i < 4; ++i) {
      int idx2 = (tid + i * 256) * 2;
      int key = idx2 >> 6, e = idx2 & 63;
      int gk = kt + key; if (gk >= S_) gk = S_ - 1;
      unsigned u = *(const unsigned int*)&v[base + (size_t)gk * D_ + e];
      vsT[(e + 0) * 32 + key] = (unsigned short)u;
      vsT[(e + 1) * 32 + key] = (unsigned short)(u >> 16);
    }
    wait_async0();
    __syncthreads();

    v8f sacc[2] = {};
    #pragma unroll
    for (int nt = 0; nt < 2; ++nt) {
      #pragma unroll
      for (int c0 = 0; c0 < 2; ++c0) {
        FragBF kfr;
        #pragma unroll
        for (int p = 0; p < 8; ++p)
          kfr.u[p] = *(const unsigned int*)
              &ks[(nt * 16 + m16) * 64 + c0 * 32 + afrag_kk(p, hf)];
        sacc[nt] = __builtin_amdgcn_wmma_f32_16x16x32_bf16(
            false, qa[c0].bf, false, kfr.bf, (short)0, sacc[nt], false, false);
      }
    }

    const bool c0v = (kt + m16) < S_;
    const bool c1v = (kt + 16 + m16) < S_;
    #pragma unroll
    for (int i = 0; i < 8; ++i) {
      float s0 = c0v ? sacc[0][i] * scale : -3.4e38f;
      float s1 = c1v ? sacc[1][i] * scale : -3.4e38f;
      float mx = fmaxf(fmaxf(s0, s1), m_run[i]);
      #pragma unroll
      for (int d = 1; d < 16; d <<= 1) mx = fmaxf(mx, __shfl_xor(mx, d, 16));
      float al = __expf(m_run[i] - mx);
      float p0 = c0v ? __expf(s0 - mx) : 0.f;
      float p1 = c1v ? __expf(s1 - mx) : 0.f;
      float rs = p0 + p1;
      #pragma unroll
      for (int d = 1; d < 16; d <<= 1) rs += __shfl_xor(rs, d, 16);
      l_run[i] = l_run[i] * al + rs;
      m_run[i] = mx;
      #pragma unroll
      for (int nt = 0; nt < 4; ++nt) oacc[nt][i] *= al;
      int row = i + 8 * hf;
      plds[pbase + row * 32 + m16]      = f2bf(p0);
      plds[pbase + row * 32 + 16 + m16] = f2bf(p1);
    }
    __syncthreads();

    FragBF pa;
    #pragma unroll
    for (int p = 0; p < 8; ++p)
      pa.u[p] = *(const unsigned int*)&plds[pbase + m16 * 32 + afrag_kk(p, hf)];
    #pragma unroll
    for (int nt = 0; nt < 4; ++nt) {
      FragBF vfr;
      #pragma unroll
      for (int p = 0; p < 8; ++p)
        vfr.u[p] = *(const unsigned int*)
            &vsT[(nt * 16 + m16) * 32 + afrag_kk(p, hf)];
      oacc[nt] = __builtin_amdgcn_wmma_f32_16x16x32_bf16(
          false, pa.bf, false, vfr.bf, (short)0, oacc[nt], false, false);
    }
    __syncthreads();
  }

  #pragma unroll
  for (int nt = 0; nt < 4; ++nt)
    #pragma unroll
    for (int i = 0; i < 8; ++i) {
      int gq = q0 + wid * 16 + i + 8 * hf;
      if (gq < S_)
        out[base + (size_t)gq * D_ + nt * 16 + m16] += oacc[nt][i] / l_run[i];
    }
}

// ---------------------------------------------------------------------------
__global__ __launch_bounds__(256) void vit_layernorm_bf(
    const float* __restrict__ x, const float* __restrict__ g,
    const float* __restrict__ b, unsigned short* __restrict__ y)
{
  __shared__ float s1[256], s2[256];
  const int row = blockIdx.x;
  const float* xr = x + (size_t)row * D_;
  float sum = 0.f, sq = 0.f;
  for (int i = threadIdx.x; i < D_; i += 256) { float v = xr[i]; sum += v; sq += v * v; }
  s1[threadIdx.x] = sum; s2[threadIdx.x] = sq;
  __syncthreads();
  for (int off = 128; off > 0; off >>= 1) {
    if (threadIdx.x < off) { s1[threadIdx.x] += s1[threadIdx.x + off]; s2[threadIdx.x] += s2[threadIdx.x + off]; }
    __syncthreads();
  }
  float mu  = s1[0] * (1.0f / D_);
  float var = s2[0] * (1.0f / D_) - mu * mu;
  float inv = rsqrtf(var + EPS_);
  unsigned short* yr = y + (size_t)row * D_;
  for (int i = threadIdx.x; i < D_; i += 256)
    yr[i] = f2bf((xr[i] - mu) * inv * g[i] + b[i]);
}

// ---------------------------------------------------------------------------
// Weight conversion: dst[n*Kpad+k] = bf16(src[k*N+n]) with zero padding.
__global__ void cvt_transpose_bf16(const float* __restrict__ src,
                                   unsigned short* __restrict__ dst,
                                   int K, int N, int Kpad, int Npad) {
  int id = blockIdx.x * 256 + threadIdx.x;
  int total = Npad * Kpad;
  if (id >= total) return;
  int n = id / Kpad, kk = id % Kpad;
  float v = (n < N && kk < K) ? src[(size_t)kk * N + n] : 0.f;
  dst[id] = f2bf(v);
}

__global__ void cvt_bf16(const float* __restrict__ src,
                         unsigned short* __restrict__ dst, int total) {
  int id = blockIdx.x * 256 + threadIdx.x;
  if (id < total) dst[id] = f2bf(src[id]);
}

// extract cls rows (padded to 128 rows, zeros beyond B_)
__global__ void cvt_cls(const float* __restrict__ outb,
                        unsigned short* __restrict__ dst) {
  int id = blockIdx.x * 256 + threadIdx.x;
  if (id >= 128 * D_) return;
  int r = id / D_, d = id % D_;
  float v = (r < B_) ? outb[(size_t)r * S_ * D_ + d] : 0.f;
  dst[id] = f2bf(v);
}

// patchify -> bf16 with K padded to INPAD_
__global__ void vit_patchify_bf(const float* __restrict__ x,
                                unsigned short* __restrict__ p) {
  int id = blockIdx.x * blockDim.x + threadIdx.x;
  const int total = B_ * NP_ * NP_ * INPAD_;
  if (id >= total) return;
  int cin = id % INPAD_;
  int bp  = id / INPAD_;
  if (cin >= IN_DIM_) { p[id] = 0; return; }
  int b   = bp / (NP_ * NP_);
  int pp  = bp % (NP_ * NP_);
  int py = pp / NP_, px = pp % NP_;
  int c   = cin / (PS_ * PS_);
  int rem = cin % (PS_ * PS_);
  int iy = rem / PS_, ix = rem % PS_;
  size_t src = (((size_t)b * C_ + c) * IMG_ + (py * PS_ + iy)) * IMG_ + (px * PS_ + ix);
  p[id] = f2bf(x[src]);
}

__global__ void vit_assemble(const float* __restrict__ proj, const float* __restrict__ cls,
                             const float* __restrict__ pos, float* __restrict__ out) {
  int id = blockIdx.x * blockDim.x + threadIdx.x;
  const int total = B_ * S_ * D_;
  if (id >= total) return;
  int d  = id % D_;
  int rs = id / D_;
  int s  = rs % S_;
  int b  = rs / S_;
  float v = (s == 0) ? cls[d]
                     : proj[((size_t)b * (NP_ * NP_) + (s - 1)) * D_ + d];
  out[id] = v + pos[(size_t)s * D_ + d];
}

__global__ __launch_bounds__(256) void vit_softmax(const float* __restrict__ logits,
                                                   float* __restrict__ out) {
  __shared__ float red[256];
  const int row = blockIdx.x;
  const float* lr = logits + (size_t)row * OUT_;
  float mx = -3.4e38f;
  for (int i = threadIdx.x; i < OUT_; i += 256) mx = fmaxf(mx, lr[i]);
  red[threadIdx.x] = mx;
  __syncthreads();
  for (int off = 128; off > 0; off >>= 1) {
    if (threadIdx.x < off) red[threadIdx.x] = fmaxf(red[threadIdx.x], red[threadIdx.x + off]);
    __syncthreads();
  }
  mx = red[0];
  __syncthreads();
  float sum = 0.f;
  for (int i = threadIdx.x; i < OUT_; i += 256) sum += __expf(lr[i] - mx);
  red[threadIdx.x] = sum;
  __syncthreads();
  for (int off = 128; off > 0; off >>= 1) {
    if (threadIdx.x < off) red[threadIdx.x] += red[threadIdx.x + off];
    __syncthreads();
  }
  float inv = 1.0f / red[0];
  float* orow = out + (size_t)row * OUT_;
  for (int i = threadIdx.x; i < OUT_; i += 256) orow[i] = __expf(lr[i] - mx) * inv;
}

// ---------------------------------------------------------------------------
extern "C" void kernel_launch(void* const* d_in, const int* in_sizes, int n_in,
                              void* d_out, int out_size, void* d_ws, size_t ws_size,
                              hipStream_t stream) {
  (void)in_sizes; (void)n_in; (void)out_size; (void)ws_size;
  const float* x     = (const float*)d_in[0];
  const float* cls   = (const float*)d_in[1];
  const float* pos   = (const float*)d_in[2];
  const float* Wmap  = (const float*)d_in[3];
  const float* bmap  = (const float*)d_in[4];
  const float* ln1_g = (const float*)d_in[5];
  const float* ln1_b = (const float*)d_in[6];
  const float* Wq    = (const float*)d_in[7];
  const float* bq    = (const float*)d_in[8];
  const float* Wk    = (const float*)d_in[9];
  const float* bk    = (const float*)d_in[10];
  const float* Wv    = (const float*)d_in[11];
  const float* bv    = (const float*)d_in[12];
  const float* ln2_g = (const float*)d_in[13];
  const float* ln2_b = (const float*)d_in[14];
  const float* W1    = (const float*)d_in[15];
  const float* b1    = (const float*)d_in[16];
  const float* W2    = (const float*)d_in[17];
  const float* b2    = (const float*)d_in[18];
  const float* Wc    = (const float*)d_in[19];
  const float* bc    = (const float*)d_in[20];

  char* ws = (char*)d_ws;
  size_t off = 0;
  auto take = [&](size_t bytes) { size_t r = off; off += (bytes + 255) & ~(size_t)255; return r; };

  float*          outb   = (float*)(ws + take((size_t)TOK_ * D_ * 4));        // 19.4 MB
  unsigned short* xnbf   = (unsigned short*)(ws + take((size_t)TOKPAD_ * D_ * 2));
  unsigned short* qbf    = (unsigned short*)(ws + take((size_t)TOKPAD_ * D_ * 2));
  unsigned short* kbf    = (unsigned short*)(ws + take((size_t)TOKPAD_ * D_ * 2));
  unsigned short* vbf    = (unsigned short*)(ws + take((size_t)TOKPAD_ * D_ * 2));
  unsigned short* hbf    = (unsigned short*)(ws + take((size_t)TOKPAD_ * MLP_ * 2)); // 39.3 MB
  unsigned short* wmapbf = (unsigned short*)(ws + take((size_t)D_ * INPAD_ * 2));
  unsigned short* wqbf   = (unsigned short*)(ws + take((size_t)L_ * NH_ * DH_ * DH_ * 2));
  unsigned short* wkbf   = (unsigned short*)(ws + take((size_t)L_ * NH_ * DH_ * DH_ * 2));
  unsigned short* wvbf   = (unsigned short*)(ws + take((size_t)L_ * NH_ * DH_ * DH_ * 2));
  unsigned short* w1bfc  = (unsigned short*)(ws + take((size_t)MLP_ * D_ * 2));      // per-layer
  unsigned short* w2bfc  = (unsigned short*)(ws + take((size_t)MLP_ * D_ * 2));      // per-layer
  unsigned short* wcbf   = (unsigned short*)(ws + take((size_t)OUTPAD_ * D_ * 2));
  unsigned short* clsbf  = (unsigned short*)(ws + take((size_t)128 * D_ * 2));
  // aliases (lifetimes disjoint):
  unsigned short* patchesbf = qbf;    // 6272*608*2 = 7.6 MB  <= qbf (9.8 MB)
  float*          proj      = (float*)hbf;  // 19.3 MB <= hbf (39.3 MB)
  float*          logits    = (float*)kbf;  // used after last attention

  const int NPATCH = B_ * NP_ * NP_;  // 6272 (multiple of 128)

  // ---- one-time weight conversions ----
  cvt_transpose_bf16<<<(D_ * INPAD_ + 255) / 256, 256, 0, stream>>>(
      Wmap, wmapbf, IN_DIM_, D_, INPAD_, D_);
  {
    int wqkv = L_ * NH_ * DH_ * DH_;
    cvt_bf16<<<(wqkv + 255) / 256, 256, 0, stream>>>(Wq, wqbf, wqkv);
    cvt_bf16<<<(wqkv + 255) / 256, 256, 0, stream>>>(Wk, wkbf, wqkv);
    cvt_bf16<<<(wqkv + 255) / 256, 256, 0, stream>>>(Wv, wvbf, wqkv);
  }
  cvt_transpose_bf16<<<(OUTPAD_ * D_ + 255) / 256, 256, 0, stream>>>(
      Wc, wcbf, D_, OUT_, D_, OUTPAD_);

  // ---- patch embedding ----
  vit_patchify_bf<<<(NPATCH * INPAD_ + 255) / 256, 256, 0, stream>>>(x, patchesbf);
  vit_wmma_gemm_bf<<<dim3(D_ / 64, NPATCH / 128, 1), 256, 0, stream>>>(
      patchesbf, INPAD_, 0, wmapbf, INPAD_, 0, bmap, 0, nullptr,
      proj, D_, 0, 0, NPATCH, D_, INPAD_, 0);
  vit_assemble<<<(TOK_ * D_ + 255) / 256, 256, 0, stream>>>(proj, cls, pos, outb);

  for (int l = 0; l < L_; ++l) {
    vit_layernorm_bf<<<TOK_, 256, 0, stream>>>(outb, ln1_g + (size_t)l * D_,
                                               ln1_b + (size_t)l * D_, xnbf);
    const unsigned short* Wt[3] = {wqbf, wkbf, wvbf};
    const float* bt[3] = {bq, bk, bv};
    unsigned short* dst[3] = {qbf, kbf, vbf};
    for (int t = 0; t < 3; ++t) {
      vit_wmma_gemm_bf<<<dim3(1, TOKPAD_ / 128, NH_), 256, 0, stream>>>(
          xnbf, D_, DH_,
          Wt[t] + (size_t)l * NH_ * DH_ * DH_, DH_, DH_ * DH_,
          bt[t] + (size_t)l * NH_ * DH_, DH_, nullptr,
          dst[t], D_, DH_, 1, TOK_, DH_, DH_, 0);
    }
    vit_attention_wmma<<<dim3((S_ + 127) / 128, NH_, B_), 256, 0, stream>>>(
        qbf, kbf, vbf, outb);

    vit_layernorm_bf<<<TOK_, 256, 0, stream>>>(outb, ln2_g + (size_t)l * D_,
                                               ln2_b + (size_t)l * D_, xnbf);
    cvt_transpose_bf16<<<(MLP_ * D_ + 255) / 256, 256, 0, stream>>>(
        W1 + (size_t)l * D_ * MLP_, w1bfc, D_, MLP_, D_, MLP_);
    vit_wmma_gemm_bf<<<dim3(MLP_ / 64, TOKPAD_ / 128, 1), 256, 0, stream>>>(
        xnbf, D_, 0, w1bfc, D_, 0, b1 + (size_t)l * MLP_, 0, nullptr,
        hbf, MLP_, 0, 1, TOK_, MLP_, D_, 1);
    cvt_transpose_bf16<<<(MLP_ * D_ + 255) / 256, 256, 0, stream>>>(
        W2 + (size_t)l * MLP_ * D_, w2bfc, MLP_, D_, MLP_, D_);
    vit_wmma_gemm_bf<<<dim3(D_ / 64, TOKPAD_ / 128, 1), 256, 0, stream>>>(
        hbf, MLP_, 0, w2bfc, MLP_, 0, b2 + (size_t)l * D_, 0, outb,
        outb, D_, 0, 0, TOK_, D_, MLP_, 0);
  }

  cvt_cls<<<(128 * D_ + 255) / 256, 256, 0, stream>>>(outb, clsbf);
  vit_wmma_gemm_bf<<<dim3(OUTPAD_ / 64, 1, 1), 256, 0, stream>>>(
      clsbf, D_, 0, wcbf, D_, 0, bc, 0, nullptr,
      logits, OUT_, 0, 0, B_, OUT_, D_, 0);
  vit_softmax<<<B_, 256, 0, stream>>>(logits, (float*)d_out);
}
